// MelDecoder_1357209666018
// MI455X (gfx1250) — compile-verified, matrix operand order checked by reading, not run on hardware
//
#include <hip/hip_runtime.h>

typedef __attribute__((ext_vector_type(16))) __bf16 v16bf;
typedef __attribute__((ext_vector_type(8)))  float  v8f;
typedef unsigned short u16;

#define B_TOT   128
#define NGROUP  8          // batch groups of 16 rows
#define NCHUNK  16         // enc chunks of 32 positions per group
#define TDEC    200
#define TENC    512
#define DH      256

__device__ __forceinline__ u16 f2bf(float f) {
    unsigned int u = __float_as_uint(f);
    unsigned int r = u + 0x7FFFu + ((u >> 16) & 1u);
    return (u16)(r >> 16);
}
__device__ __forceinline__ float bf2f(u16 h) {
    return __uint_as_float(((unsigned int)h) << 16);
}
__device__ __forceinline__ float sigmoid_f(float x) { return 1.0f / (1.0f + __expf(-x)); }
__device__ __forceinline__ float tanh_f(float x) {
    float e = __expf(-2.0f * x);
    return (1.0f - e) / (1.0f + e);
}

// ---------------------------------------------------------------------------
// 16xN = (16xK) @ W(N,:)^T WMMA GEMM. A in LDS bf16 (ld = lda), W global bf16
// rows of length wstride (K consumed, K % 32 == 0). Accumulator init: bf16
// global tile (NT load) or f32 bias or 0. Epilogue: optional relu, f32 LDS,
// bf16 LDS, f32 global (NT), bf16 global (NT).
// ---------------------------------------------------------------------------
__device__ void wgemm(const u16* __restrict__ A, int lda,
                      const u16* __restrict__ W, int K, int wstride, int N,
                      const float* __restrict__ bias, int relu,
                      const u16* __restrict__ initB, int ldi,
                      float* fOut, int ldf,
                      u16* bOut, int ldb,
                      float* gOut, int gstride,
                      u16* gOutB, int gstrideB) {
    const int lane = threadIdx.x & 31, wave = threadIdx.x >> 5;
    const int lo = lane & 15, hi = lane >> 4;
    const int ntiles = N >> 4;
    for (int nt = wave; nt < ntiles; nt += 8) {
        const int ncol = nt * 16 + lo;
        v8f c;
        if (initB) {
            #pragma unroll
            for (int r = 0; r < 8; ++r)
                c[r] = bf2f(__builtin_nontemporal_load(initB + (size_t)(r + hi * 8) * ldi + ncol));
        } else {
            float bv = bias ? bias[ncol] : 0.0f;
            #pragma unroll
            for (int r = 0; r < 8; ++r) c[r] = bv;
        }
        const u16* wrow = W + (size_t)ncol * wstride + hi * 16;
        const u16* arow = A + lo * lda + hi * 16;
        for (int k0 = 0; k0 < K; k0 += 32) {
            v16bf a = *(const v16bf*)(arow + k0);
            v16bf b = *(const v16bf*)(wrow + k0);
            c = __builtin_amdgcn_wmma_f32_16x16x32_bf16(false, a, false, b,
                                                        (short)0, c, false, false);
        }
        #pragma unroll
        for (int r = 0; r < 8; ++r) {
            float x = c[r];
            if (relu) x = fmaxf(x, 0.0f);
            int m = r + hi * 8, col = nt * 16 + lo;
            if (fOut) fOut[m * ldf + col] = x;
            if (bOut) bOut[m * ldb + col] = f2bf(x);
            if (gOut) __builtin_nontemporal_store(x, gOut + (size_t)m * gstride + col);
            if (gOutB) __builtin_nontemporal_store(f2bf(x), gOutB + (size_t)m * gstrideB + col);
        }
    }
}

// ---------------------------------------------------------------------------
// Fused GRU: gh = h_prev @ Whh^T + bhh (3 gate sections, 3 accumulators per
// d-tile) combined in registers with precomputed gi (LDS f32 16x768); updates
// hState f32 in place; optional bf16 copy of new h.
// ---------------------------------------------------------------------------
__device__ void gru_combine(const u16* __restrict__ Hb, int ldh,
                            const u16* __restrict__ Whh,
                            const float* __restrict__ bhh,
                            const float* __restrict__ gi,
                            float* hState,
                            u16* hOutB, int ldb) {
    const int lane = threadIdx.x & 31, wave = threadIdx.x >> 5;
    const int lo = lane & 15, hi = lane >> 4;
    for (int dt = wave; dt < 16; dt += 8) {
        const int dcol = dt * 16 + lo;
        v8f cr, cz, cn;
        float br_ = bhh[dcol], bz_ = bhh[256 + dcol], bn_ = bhh[512 + dcol];
        #pragma unroll
        for (int r = 0; r < 8; ++r) { cr[r] = br_; cz[r] = bz_; cn[r] = bn_; }
        const u16* wr = Whh + (size_t)(dcol)       * 256 + hi * 16;
        const u16* wz = Whh + (size_t)(256 + dcol) * 256 + hi * 16;
        const u16* wn = Whh + (size_t)(512 + dcol) * 256 + hi * 16;
        const u16* ar = Hb + lo * ldh + hi * 16;
        for (int k0 = 0; k0 < 256; k0 += 32) {
            v16bf a  = *(const v16bf*)(ar + k0);
            v16bf b0 = *(const v16bf*)(wr + k0);
            v16bf b1 = *(const v16bf*)(wz + k0);
            v16bf b2 = *(const v16bf*)(wn + k0);
            cr = __builtin_amdgcn_wmma_f32_16x16x32_bf16(false, a, false, b0, (short)0, cr, false, false);
            cz = __builtin_amdgcn_wmma_f32_16x16x32_bf16(false, a, false, b1, (short)0, cz, false, false);
            cn = __builtin_amdgcn_wmma_f32_16x16x32_bf16(false, a, false, b2, (short)0, cn, false, false);
        }
        #pragma unroll
        for (int r = 0; r < 8; ++r) {
            int m = r + hi * 8, d = dt * 16 + lo;
            float gir = gi[m * 768 + d];
            float giz = gi[m * 768 + 256 + d];
            float gin = gi[m * 768 + 512 + d];
            float rr = sigmoid_f(gir + cr[r]);
            float zz = sigmoid_f(giz + cz[r]);
            float nn = tanh_f(gin + rr * cn[r]);
            float ho = hState[m * 256 + d];
            float hn = (1.0f - zz) * nn + zz * ho;
            hState[m * 256 + d] = hn;
            if (hOutB) hOutB[m * ldb + d] = f2bf(hn);
        }
    }
}

// --------------------------- prep kernels ----------------------------------
__global__ void k_zero(int* p, int n) {
    int i = blockIdx.x * blockDim.x + threadIdx.x;
    if (i < n) p[i] = 0;
}

__global__ void k_cvt(u16* dst, const float* src, int N, int K, int Kp) {
    int i = blockIdx.x * 256 + threadIdx.x;
    if (i >= N * Kp) return;
    int n = i / Kp, k = i % Kp;
    dst[i] = f2bf(k < K ? src[n * K + k] : 0.0f);
}

// processed_mem(bf16) = encoder_outputs @ Wm^T ; (128*512,256)x(256,256)
__global__ __launch_bounds__(256) void k_pm(const float* __restrict__ enc,
                                            const u16* __restrict__ WmB,
                                            u16* __restrict__ pm) {
    __shared__ __align__(32) u16 sA[16 * 256];
    const int blk = blockIdx.x;
    const int tid = threadIdx.x;
    {
        int r = tid >> 4, c0 = (tid & 15) * 16;
        const float* src = enc + ((size_t)(blk * 16 + r)) * 256 + c0;
        #pragma unroll
        for (int j = 0; j < 16; ++j) sA[r * 256 + c0 + j] = f2bf(src[j]);
    }
    __syncthreads();
    const int lane = tid & 31, wave = tid >> 5, lo = lane & 15, hi = lane >> 4;
    for (int nt = wave; nt < 16; nt += 8) {
        v8f c = {};
        const u16* wrow = WmB + (size_t)(nt * 16 + lo) * 256 + hi * 16;
        const u16* arow = sA + lo * 256 + hi * 16;
        for (int k0 = 0; k0 < 256; k0 += 32) {
            v16bf a = *(const v16bf*)(arow + k0);
            v16bf b = *(const v16bf*)(wrow + k0);
            c = __builtin_amdgcn_wmma_f32_16x16x32_bf16(false, a, false, b, (short)0, c, false, false);
        }
        #pragma unroll
        for (int r = 0; r < 8; ++r) {
            int row = blk * 16 + r + hi * 8;
            pm[(size_t)row * 256 + nt * 16 + lo] = f2bf(c[r]);
        }
    }
}

// Precompute (input-only part of the recurrence, fully parallel over steps):
// preG[t][b][768] = relu(relu(x_shift@W1^T+b1)@W2^T+b2) @ WihA[:, :128]^T + bih_a
__global__ __launch_bounds__(256) void k_pre(const float* __restrict__ xin,
        const u16* __restrict__ W1p, const float* __restrict__ b1,
        const u16* __restrict__ W2p, const float* __restrict__ b2,
        const u16* __restrict__ WihA, const float* __restrict__ bih_a,
        u16* __restrict__ preG) {
    __shared__ __align__(32) u16 sX[16 * 416];
    __shared__ __align__(32) u16 sB1[16 * 256];
    __shared__ __align__(32) u16 sP[16 * 128];
    const int blk = blockIdx.x;            // 1600 = 200 steps * 8 groups
    const int t = blk >> 3, gg = blk & 7;
    const int tid = threadIdx.x;
    {
        int row = tid >> 4, cb = (tid & 15) * 26;
        #pragma unroll
        for (int j = 0; j < 26; ++j) {
            int col = cb + j;
            float x = 0.0f;
            if (col < 400 && t > 0)
                x = xin[(size_t)(gg * 16 + row) * 80000 + (t - 1) * 400 + col];
            sX[row * 416 + col] = f2bf(x);
        }
    }
    __syncthreads();
    wgemm(sX, 416, W1p, 416, 416, 256, b1, 1, nullptr, 0,
          nullptr, 0, sB1, 256, nullptr, 0, nullptr, 0);
    __syncthreads();
    wgemm(sB1, 256, W2p, 256, 256, 128, b2, 1, nullptr, 0,
          nullptr, 0, sP, 128, nullptr, 0, nullptr, 0);
    __syncthreads();
    wgemm(sP, 128, WihA, 128, 384, 768, bih_a, 0, nullptr, 0,
          nullptr, 0, nullptr, 0, nullptr, 0,
          preG + (size_t)t * 128 * 768 + (size_t)(gg * 16) * 768, 768);
}

// --------------------------- main persistent kernel ------------------------
struct Params {
    const float *enc, *xin;
    const float *bhh_a, *v, *bp, *bih_d1, *bhh_d1, *bih_d2, *bhh_d2, *bmel;
    const u16 *WihA, *WhhA, *Wq, *Wp, *Wd1i, *Wd1h, *Wd2i, *Wd2h, *Wmel;
    const u16 *pm, *preG;
    float *qbuf, *stats, *rbuf, *pctx, *out, *aligns;
    int *bar;
};

__device__ void gbar(int* cnt, int* gen, int n) {
    __syncthreads();
    if (threadIdx.x == 0) {
        __threadfence();
        int g = atomicAdd(gen, 0);
        if (atomicAdd(cnt, 1) == n - 1) {
            atomicExch(cnt, 0);
            __threadfence();
            atomicAdd(gen, 1);
        } else {
            while (atomicAdd(gen, 0) == g) __builtin_amdgcn_s_sleep(2);
        }
        __threadfence();
    }
    __syncthreads();
}

__global__ __launch_bounds__(256) void k_decode(Params P) {
    // LDS ~152KB -> one WG per WGP (320KB available)
    __shared__ __align__(32) float sG[16 * 768];     // gates; low floats double as s_l/s_w
    __shared__ __align__(32) float s_hA[16 * 256];
    __shared__ __align__(32) float s_h1[16 * 256];
    __shared__ __align__(32) float s_h2[16 * 256];
    __shared__ __align__(32) float s_ctx[16 * 256];
    __shared__ __align__(32) float s_q[16 * 256];    // q; reused as dec_in accumulator
    __shared__ __align__(32) u16 sA[16 * 512];       // bf16 A staging (ctx / attn_h+ctx concat)
    __shared__ __align__(32) u16 sB[16 * 256];       // bf16 intermediate operand

    float* s_l = sG;            // 16x32 exp(logit) values (attention phase only)
    float* s_w = sG + 512;      // 16x16 merge weights   (master, attention phase)

    const int g = blockIdx.x >> 4;
    const int c = blockIdx.x & 15;
    const bool master = (c == 0);
    const int tid = threadIdx.x;
    const int lane = tid & 31, wave = tid >> 5;
    int* bcnt = P.bar + g * 16;
    int* bgen = bcnt + 1;

    // per-lane invariants for the attention inner loop
    float vreg[8];
    #pragma unroll
    for (int j = 0; j < 8; ++j) vreg[j] = P.v[lane * 8 + j];
    const u16* pmB = P.pm + ((size_t)(g * 16) * TENC + c * 32) * 256;

    if (master) {
        int i0 = tid * 16;
        #pragma unroll
        for (int j = 0; j < 16; ++j) {
            s_hA[i0 + j] = 0.0f; s_h1[i0 + j] = 0.0f;
            s_h2[i0 + j] = 0.0f; s_ctx[i0 + j] = 0.0f;
        }
    }
    __syncthreads();

    for (int step = 0; step < TDEC; ++step) {
        if (master) {
            {   // stage ctx (A operand) and attn_h (combine operand) as bf16
                int i0 = tid * 16;
                #pragma unroll
                for (int j = 0; j < 16; ++j) {
                    sA[i0 + j] = f2bf(s_ctx[i0 + j]);
                    sB[i0 + j] = f2bf(s_hA[i0 + j]);
                }
            }
            __syncthreads();
            // gi = preG[step] (prenet+input part, precomputed) + ctx @ WihA[:,128:]^T
            wgemm(sA, 256, P.WihA + 128, 256, 384, 768, nullptr, 0,
                  P.preG + (size_t)step * 128 * 768 + (size_t)(g * 16) * 768, 768,
                  sG, 768, nullptr, 0, nullptr, 0, nullptr, 0);
            __syncthreads();
            gru_combine(sB, 256, P.WhhA, P.bhh_a, sG, s_hA, sA, 512); // new attn_h bf16 -> sA[:,0:256]
            __syncthreads();
            wgemm(sA, 512, P.Wq, 256, 256, 256, nullptr, 0, nullptr, 0,
                  s_q, 256, nullptr, 0, nullptr, 0, nullptr, 0);
            __syncthreads();
            {   int i0 = tid * 16;
                #pragma unroll
                for (int j = 0; j < 16; ++j) P.qbuf[g * 4096 + i0 + j] = s_q[i0 + j];
            }
        }
        gbar(bcnt, bgen, NCHUNK);                          // B1: q published

        // ================= attention (all 16 WGs, 32 enc pos each) =========
        if (!master) {
            int i0 = tid * 16;
            #pragma unroll
            for (int j = 0; j < 16; ++j) s_q[i0 + j] = P.qbuf[g * 4096 + i0 + j];
        }
        __syncthreads();
        for (int idx = wave; idx < 512; idx += 8) {        // 16 rows x 32 pos
            int m = idx & 15, t = idx >> 4;
            const u16* pr = pmB + ((size_t)m * TENC + t) * 256 + lane * 8;
            uint4 pv = *(const uint4*)pr;                  // 8 bf16 per lane, one b128
            float pmv[8];
            pmv[0] = __uint_as_float(pv.x << 16); pmv[1] = __uint_as_float(pv.x & 0xFFFF0000u);
            pmv[2] = __uint_as_float(pv.y << 16); pmv[3] = __uint_as_float(pv.y & 0xFFFF0000u);
            pmv[4] = __uint_as_float(pv.z << 16); pmv[5] = __uint_as_float(pv.z & 0xFFFF0000u);
            pmv[6] = __uint_as_float(pv.w << 16); pmv[7] = __uint_as_float(pv.w & 0xFFFF0000u);
            float acc = 0.0f;
            int qb = m * 256 + lane * 8;
            #pragma unroll
            for (int j = 0; j < 8; ++j)
                acc += vreg[j] * tanh_f(s_q[qb + j] + pmv[j]);
            #pragma unroll
            for (int off = 16; off > 0; off >>= 1) acc += __shfl_xor(acc, off, 32);
            if (lane == 0) s_l[m * 32 + t] = acc;
        }
        __syncthreads();
        if (tid < 16) {                                    // chunk-local softmax stats
            int m = tid;
            float mx = -1e30f;
            for (int t = 0; t < 32; ++t) mx = fmaxf(mx, s_l[m * 32 + t]);
            float s = 0.0f;
            for (int t = 0; t < 32; ++t) {
                float e = __expf(s_l[m * 32 + t] - mx);
                s_l[m * 32 + t] = e; s += e;
            }
            P.stats[((g * 16 + c) * 16 + m) * 2 + 0] = mx;
            P.stats[((g * 16 + c) * 16 + m) * 2 + 1] = s;
        }
        __syncthreads();
        {   // partial ctx over this chunk
            int m = tid >> 4, d0 = (tid & 15) * 16;
            float acc[16];
            #pragma unroll
            for (int j = 0; j < 16; ++j) acc[j] = 0.0f;
            const float* eb = P.enc + ((size_t)(g * 16 + m) * TENC + c * 32) * 256 + d0;
            for (int t = 0; t < 32; ++t) {
                float e = s_l[m * 32 + t];
                const float* p = eb + (size_t)t * 256;
                #pragma unroll
                for (int j = 0; j < 16; ++j) acc[j] += e * p[j];
            }
            float* dst = P.pctx + ((size_t)((g * 16 + c) * 16 + m)) * 256 + d0;
            #pragma unroll
            for (int j = 0; j < 16; ++j) dst[j] = acc[j];
        }
        __threadfence();
        gbar(bcnt, bgen, NCHUNK);                          // B2: partials published

        if (master) {
            if (tid < 16) {                                // merge softmax stats
                int m = tid;
                float mx = -1e30f;
                for (int i = 0; i < 16; ++i)
                    mx = fmaxf(mx, P.stats[((g * 16 + i) * 16 + m) * 2 + 0]);
                float S = 0.0f;
                for (int i = 0; i < 16; ++i)
                    S += P.stats[((g * 16 + i) * 16 + m) * 2 + 1] *
                         __expf(P.stats[((g * 16 + i) * 16 + m) * 2 + 0] - mx);
                float inv = 1.0f / S;
                for (int i = 0; i < 16; ++i) {
                    float r = __expf(P.stats[((g * 16 + i) * 16 + m) * 2 + 0] - mx) * inv;
                    s_w[i * 16 + m] = r;
                    P.rbuf[(g * 16 + i) * 16 + m] = r;
                }
            }
            __syncthreads();
            {   // ctx = sum_i w_i * pctx_i
                int m = tid >> 4, d0 = (tid & 15) * 16;
                float acc[16];
                #pragma unroll
                for (int j = 0; j < 16; ++j) acc[j] = 0.0f;
                for (int i = 0; i < 16; ++i) {
                    float w = s_w[i * 16 + m];
                    const float* p = P.pctx + ((size_t)((g * 16 + i) * 16 + m)) * 256 + d0;
                    #pragma unroll
                    for (int j = 0; j < 16; ++j) acc[j] += w * p[j];
                }
                #pragma unroll
                for (int j = 0; j < 16; ++j) s_ctx[m * 256 + d0 + j] = acc[j];
            }
            __threadfence();
        }
        gbar(bcnt, bgen, NCHUNK);                          // B3: rescale factors ready

        // normalized alignments -> d_out (streaming, non-temporal)
        for (int i = tid; i < 512; i += 256) {
            int m = i >> 5, t = i & 31;
            float r = P.rbuf[(g * 16 + c) * 16 + m];
            __builtin_nontemporal_store(
                s_l[m * 32 + t] * r,
                P.aligns + ((size_t)(g * 16 + m) * TDEC + step) * TENC + c * 32 + t);
        }
        __syncthreads();

        if (master) {
            // ---- decoder: dec_in = [attn_h, ctx] @ Wp^T + bp ----
            {   int m = tid >> 4, d0 = (tid & 15) * 16;   // attn_h bf16 already in sA[:,0:256]
                #pragma unroll
                for (int j = 0; j < 16; ++j)
                    sA[m * 512 + 256 + d0 + j] = f2bf(s_ctx[m * 256 + d0 + j]);
            }
            __syncthreads();
            wgemm(sA, 512, P.Wp, 512, 512, 256, P.bp, 0, nullptr, 0,
                  s_q, 256, sB, 256, nullptr, 0, nullptr, 0);
            __syncthreads();
            // ---- GRU d1 ----
            wgemm(sB, 256, P.Wd1i, 256, 256, 768, P.bih_d1, 0, nullptr, 0,
                  sG, 768, nullptr, 0, nullptr, 0, nullptr, 0);
            __syncthreads();
            {   int i0 = tid * 16;
                #pragma unroll
                for (int j = 0; j < 16; ++j) sB[i0 + j] = f2bf(s_h1[i0 + j]);
            }
            __syncthreads();
            gru_combine(sB, 256, P.Wd1h, P.bhh_d1, sG, s_h1, nullptr, 0);
            __syncthreads();
            {   int i0 = tid * 16;                        // dec_in += h1
                #pragma unroll
                for (int j = 0; j < 16; ++j) {
                    float v2 = s_q[i0 + j] + s_h1[i0 + j];
                    s_q[i0 + j] = v2; sB[i0 + j] = f2bf(v2);
                }
            }
            __syncthreads();
            // ---- GRU d2 ----
            wgemm(sB, 256, P.Wd2i, 256, 256, 768, P.bih_d2, 0, nullptr, 0,
                  sG, 768, nullptr, 0, nullptr, 0, nullptr, 0);
            __syncthreads();
            {   int i0 = tid * 16;
                #pragma unroll
                for (int j = 0; j < 16; ++j) sB[i0 + j] = f2bf(s_h2[i0 + j]);
            }
            __syncthreads();
            gru_combine(sB, 256, P.Wd2h, P.bhh_d2, sG, s_h2, nullptr, 0);
            __syncthreads();
            {   int i0 = tid * 16;                        // dec_in += h2
                #pragma unroll
                for (int j = 0; j < 16; ++j) {
                    float v2 = s_q[i0 + j] + s_h2[i0 + j];
                    s_q[i0 + j] = v2; sB[i0 + j] = f2bf(v2);
                }
            }
            __syncthreads();
            // ---- mel projection straight to global (NT) ----
            wgemm(sB, 256, P.Wmel, 256, 256, 400, P.bmel, 0, nullptr, 0,
                  nullptr, 0, nullptr, 0,
                  P.out + (size_t)(g * 16) * (TDEC * 400) + (size_t)step * 400, TDEC * 400,
                  nullptr, 0);
            __syncthreads();
        }
    }
}

// ---------------------------------------------------------------------------
extern "C" void kernel_launch(void* const* d_in, const int* in_sizes, int n_in,
                              void* d_out, int out_size, void* d_ws, size_t ws_size,
                              hipStream_t stream) {
    (void)in_sizes; (void)n_in; (void)out_size; (void)ws_size;
    const float* enc = (const float*)d_in[0];
    const float* xin = (const float*)d_in[1];

    char* base = (char*)d_ws;
    int*   bar   = (int*)base;                                   //   1 KB
    float* qbuf  = (float*)(base + 1024);                        // 128 KB
    float* stats = qbuf + NGROUP * 16 * 256;                     //  16 KB
    float* rbuf  = stats + NGROUP * NCHUNK * 16 * 2;             //   8 KB
    float* pctx  = rbuf + NGROUP * NCHUNK * 16;                  //   2 MB
    u16*   wts   = (u16*)(pctx + NGROUP * NCHUNK * 16 * 256);

    u16* W1p  = wts;              // 256 x 416
    u16* W2p  = W1p  + 256 * 416; // 128 x 256
    u16* WihA = W2p  + 128 * 256; // 768 x 384
    u16* WhhA = WihA + 768 * 384; // 768 x 256
    u16* Wq   = WhhA + 768 * 256; // 256 x 256
    u16* Wp   = Wq   + 256 * 256; // 256 x 512
    u16* Wd1i = Wp   + 256 * 512; // 768 x 256
    u16* Wd1h = Wd1i + 768 * 256;
    u16* Wd2i = Wd1h + 768 * 256;
    u16* Wd2h = Wd2i + 768 * 256;
    u16* Wmel = Wd2h + 768 * 256; // 400 x 256
    u16* WmB  = Wmel + 400 * 256; // 256 x 256
    u16* pm   = WmB  + 256 * 256;                 // 128*512 x 256 bf16 (32 MB)
    u16* preG = pm + (size_t)B_TOT * TENC * 256;  // 200 x 128 x 768 bf16 (39 MB)

    k_zero<<<1, 256, 0, stream>>>(bar, 256);

    auto cvt = [&](u16* dst, int src_idx, int N, int K, int Kp) {
        int tot = N * Kp;
        k_cvt<<<(tot + 255) / 256, 256, 0, stream>>>(dst, (const float*)d_in[src_idx], N, K, Kp);
    };
    cvt(W1p,  2, 256, 400, 416);
    cvt(W2p,  4, 128, 256, 256);
    cvt(WihA, 6, 768, 384, 384);
    cvt(WhhA, 7, 768, 256, 256);
    cvt(Wq,  10, 256, 256, 256);
    cvt(WmB, 12, 256, 256, 256);
    cvt(Wp,  13, 256, 512, 512);
    cvt(Wd1i,15, 768, 256, 256);
    cvt(Wd1h,16, 768, 256, 256);
    cvt(Wd2i,19, 768, 256, 256);
    cvt(Wd2h,20, 768, 256, 256);
    cvt(Wmel,23, 400, 256, 256);

    k_pm<<<(B_TOT * TENC) / 16, 256, 0, stream>>>(enc, WmB, pm);
    k_pre<<<TDEC * NGROUP, 256, 0, stream>>>(xin, W1p, (const float*)d_in[3],
                                             W2p, (const float*)d_in[5],
                                             WihA, (const float*)d_in[8], preG);

    Params P;
    P.enc = enc; P.xin = xin;
    P.bhh_a = (const float*)d_in[9];
    P.v = (const float*)d_in[11];   P.bp = (const float*)d_in[14];
    P.bih_d1 = (const float*)d_in[17]; P.bhh_d1 = (const float*)d_in[18];
    P.bih_d2 = (const float*)d_in[21]; P.bhh_d2 = (const float*)d_in[22];
    P.bmel = (const float*)d_in[24];
    P.WihA = WihA; P.WhhA = WhhA; P.Wq = Wq; P.Wp = Wp;
    P.Wd1i = Wd1i; P.Wd1h = Wd1h; P.Wd2i = Wd2i; P.Wd2h = Wd2h; P.Wmel = Wmel;
    P.pm = pm; P.preG = preG;
    P.qbuf = qbuf; P.stats = stats; P.rbuf = rbuf; P.pctx = pctx;
    P.out = (float*)d_out;
    P.aligns = (float*)d_out + (size_t)B_TOT * TDEC * 400;
    P.bar = bar;

    k_decode<<<NGROUP * NCHUNK, 256, 0, stream>>>(P);
}
